// DeformLayer_76871324663972
// MI455X (gfx1250) — compile-verified, matrix operand order checked by reading, not run on hardware
//
#include <hip/hip_runtime.h>
#include <hip/hip_bf16.h>
#include <math.h>

typedef __attribute__((ext_vector_type(16))) _Float16 v16h;
typedef __attribute__((ext_vector_type(8)))  _Float16 v8h;
typedef __attribute__((ext_vector_type(8)))  float    v8f;

#define BATCH 4
#define CIN   256
#define HH    64
#define WW    64
#define HW    (HH*WW)        // 4096
#define OCH   256
#define KTAP  9
#define KDCN  (KTAP*CIN)     // 2304
#define KUP   (4*CIN)        // 1024
#define HO    128
#define WO    128
#define EPSBN 1e-5f
#define LDSP  40             // padded row length (halves): 16B-aligned fragment loads

#if defined(__gfx1250__) && __has_builtin(__builtin_amdgcn_tensor_load_to_lds)
#define HAVE_TDM 1
#else
#define HAVE_TDM 0
#endif

#if HAVE_TDM
typedef __attribute__((ext_vector_type(4))) unsigned int v4u;
typedef __attribute__((ext_vector_type(4))) int          v4i;
typedef __attribute__((ext_vector_type(8))) int          v8i;

// TDM: load a 2D f16 tile [128 rows x 32 halves], row stride ktot halves,
// into LDS with +16B padding after each 64B row (=> LDSP*2 = 80B LDS row stride).
// clang-23 arity: (v4u g0, v8i g1, v4i g2, v4i g3, v8i g4, i32 cpol)
__device__ __forceinline__ void tdm_load_a_tile(const _Float16* gsrc, unsigned lds_off,
                                                unsigned ktot) {
  unsigned long long ga = (unsigned long long)(const void*)gsrc;
  v4u g0;
  g0[0] = 1u;                                            // count=1 (valid user D#)
  g0[1] = lds_off;                                       // LDS byte address
  g0[2] = (unsigned)ga;                                  // global_addr[31:0]
  g0[3] = (unsigned)((ga >> 32) & 0x1FFFFFFu) | (2u << 30); // addr[56:32] | type=2
  v8i g1;
  g1[0] = (1 << 16)                                      // data_size = 2B
        | (1 << 20)                                      // pad_enable
        | (3 << 22)                                      // pad_interval: 16 DWORDs
        | (3 << 25);                                     // pad_amount: 4 DWORDs
  g1[1] = (int)((ktot & 0xFFFFu) << 16);                 // tensor_dim0[15:0]
  g1[2] = (int)(((ktot >> 16) & 0xFFFFu) | (256u << 16));// dim0[31:16] | dim1[15:0]=256
  g1[3] = (int)(32u << 16);                              // dim1[31:16]=0 | tile_dim0=32
  g1[4] = 128;                                           // tile_dim1=128 | tile_dim2=0
  g1[5] = (int)ktot;                                     // tensor_dim0_stride[31:0]
  g1[6] = 0;                                             // stride0[47:32] | stride1[15:0]
  g1[7] = 0;
  v4i z4 = {0, 0, 0, 0};
  v8i z8 = {0, 0, 0, 0, 0, 0, 0, 0};
  __builtin_amdgcn_tensor_load_to_lds(g0, g1, z4, z4, z8, 0);
}
#endif

// ---------------- 1) offset conv: 3x3, 256 -> 27 channels, pad 1 ----------------
__global__ void dl_offset_conv(const float* __restrict__ x,
                               const float* __restrict__ w_off,
                               const float* __restrict__ b_off,
                               float* __restrict__ om) {
  int blk = blockIdx.x;
  int h  = blk % HH;
  int oc = (blk / HH) % 27;
  int b  = blk / (HH * 27);
  int w  = threadIdx.x;
  __shared__ float wsh[KDCN];
  for (int i = threadIdx.x; i < KDCN; i += 64) wsh[i] = w_off[oc * KDCN + i];
  __syncthreads();
  float acc = b_off[oc];
  const float* xb = x + (long)b * CIN * HW;
  for (int c = 0; c < CIN; ++c) {
    const float* xc = xb + c * HW;
    const float* wr = wsh + c * 9;
#pragma unroll
    for (int ky = 0; ky < 3; ++ky) {
      int yy = h + ky - 1;
      if (yy < 0 || yy >= HH) continue;
#pragma unroll
      for (int kx = 0; kx < 3; ++kx) {
        int xx = w + kx - 1;
        if (xx < 0 || xx >= WW) continue;
        acc = fmaf(xc[yy * WW + xx], wr[ky * 3 + kx], acc);
      }
    }
  }
  om[((b * 27 + oc) * HW) + h * WW + w] = acc;
}

// ---------------- 2) bilinear coords + corner weights (mask folded in) ----------------
__global__ void dl_coords(const float* __restrict__ om,
                          int4* __restrict__ ci, float4* __restrict__ cw) {
  int g = blockIdx.x * blockDim.x + threadIdx.x;
  if (g >= BATCH * KTAP * HW) return;
  int p = g % HW;
  int t = (g / HW) % KTAP;
  int b = g / (HW * KTAP);
  int h = p / WW, w = p % WW;
  const float* omb = om + (long)b * 27 * HW;
  float offx = omb[t * HW + p];
  float offy = omb[(9 + t) * HW + p];
  float m    = omb[(18 + t) * HW + p];
  m = 1.0f / (1.0f + __expf(-m));
  float sy = offy + (float)(t / 3 - 1) + (float)h;
  float sx = offx + (float)(t % 3 - 1) + (float)w;
  float y0f = floorf(sy), x0f = floorf(sx);
  float ly = sy - y0f, lx = sx - x0f;
  int y0 = (int)y0f, x0 = (int)x0f;
  int y1 = y0 + 1,  x1 = x0 + 1;
  float vy0 = (y0 >= 0 && y0 < HH) ? 1.f : 0.f;
  float vy1 = (y1 >= 0 && y1 < HH) ? 1.f : 0.f;
  float vx0 = (x0 >= 0 && x0 < WW) ? 1.f : 0.f;
  float vx1 = (x1 >= 0 && x1 < WW) ? 1.f : 0.f;
  int y0c = min(max(y0, 0), HH - 1), y1c = min(max(y1, 0), HH - 1);
  int x0c = min(max(x0, 0), WW - 1), x1c = min(max(x1, 0), WW - 1);
  ci[g] = make_int4(y0c * WW + x0c, y0c * WW + x1c, y1c * WW + x0c, y1c * WW + x1c);
  cw[g] = make_float4((1.f - ly) * (1.f - lx) * vy0 * vx0 * m,
                      (1.f - ly) * lx         * vy0 * vx1 * m,
                      ly * (1.f - lx)         * vy1 * vx0 * m,
                      ly * lx                 * vy1 * vx1 * m);
}

// ---------------- 3) im2col sampler -> f16 val[b][tap*256+c][p] ----------------
__global__ void dl_sample(const float* __restrict__ x,
                          const int4* __restrict__ ci,
                          const float4* __restrict__ cw,
                          _Float16* __restrict__ val) {
  long g = (long)blockIdx.x * blockDim.x + threadIdx.x;
  int p    = (int)(g % HW);
  int kidx = (int)((g / HW) % KDCN);
  int b    = (int)(g / ((long)HW * KDCN));
  int t = kidx / CIN, c = kidx % CIN;
  int cg = (b * KTAP + t) * HW + p;
  int4   ix = ci[cg];
  float4 wt = cw[cg];
  const float* xc = x + (long)(b * CIN + c) * HW;
  float v = wt.x * xc[ix.x] + wt.y * xc[ix.y] + wt.z * xc[ix.z] + wt.w * xc[ix.w];
  val[g] = (_Float16)v;
}

// ---------------- weight converts to f16 (GEMM A layout) ----------------
__global__ void dl_cvt_wdcn(const float* __restrict__ w, _Float16* __restrict__ A) {
  int g = blockIdx.x * 256 + threadIdx.x;
  if (g >= OCH * KDCN) return;
  int o = g / KDCN, kidx = g % KDCN;
  int t = kidx / CIN, c = kidx % CIN;
  A[g] = (_Float16)w[(o * CIN + c) * 9 + t];
}

__global__ void dl_cvt_wup(const float* __restrict__ w, _Float16* __restrict__ A) {
  int g = blockIdx.x * 256 + threadIdx.x;
  if (g >= 4 * OCH * KUP) return;
  int par  = g / (OCH * KUP);
  int oc   = (g / KUP) % OCH;
  int kidx = g % KUP;
  int tau = kidx / CIN, ic = kidx % CIN;
  int a = par >> 1, bb = par & 1;
  int s = tau >> 1, tt = tau & 1;
  int kh = a + 2 * s, kw = bb + 2 * tt;
  A[g] = (_Float16)w[((ic * OCH + oc) * 4 + (3 - kh)) * 4 + (3 - kw)];
}

// ---------------- 4) DCN GEMM: [256 x 2304] x [2304 x 4096] per batch ----------------
// Block tile 128(M) x 64(N); 8 waves as 4x2 of 32x32 wave tiles (4 WMMA / wave / k-step).
__global__ void __launch_bounds__(256)
dl_gemm_dcn(const _Float16* __restrict__ A,      // [256][2304] f16
            const _Float16* __restrict__ val,    // [B][2304][4096] f16
            float* __restrict__ y) {             // [B][256][4096] f32 (raw)
  __shared__ _Float16 Alds[128 * LDSP];
  __shared__ _Float16 Blds[64 * LDSP];
  int blk = blockIdx.x;
  int nt = blk % 64;
  int mt = (blk / 64) % 2;
  int b  = blk / 128;
  int p0 = nt * 64, m0 = mt * 128;
  int tid = threadIdx.x;
  int lane = tid & 31, wid = tid >> 5;
  int hl = lane & 15, hi = lane >> 4;
  int wm = wid & 3, wn = wid >> 2;                 // 4 M-subtiles(32) x 2 N-subtiles(32)
  v8f acc[2][2] = {};
  const _Float16* Bg = val + (long)b * KDCN * HW;
  int bkk = tid & 31, bng = tid >> 5;              // B staging: k-row, group of 8 cols
#if HAVE_TDM
  unsigned ldsA = (unsigned)(unsigned long long)(const void*)&Alds[0];
#endif
  for (int k0 = 0; k0 < KDCN; k0 += 32) {
#if !HAVE_TDM
    v8h av0 = *(const v8h*)(A + (long)(m0 + (tid >> 2)) * KDCN + k0 + (tid & 3) * 8);
    v8h av1 = *(const v8h*)(A + (long)(m0 + ((tid + 256) >> 2)) * KDCN + k0 + (tid & 3) * 8);
#endif
    v8h bv = *(const v8h*)(Bg + (long)(k0 + bkk) * HW + p0 + bng * 8);
    __syncthreads();                               // previous tile fully consumed
#if HAVE_TDM
    if (tid == 0) tdm_load_a_tile(A + (long)m0 * KDCN + k0, ldsA, KDCN);
#else
    *(v8h*)(Alds + (tid >> 2) * LDSP + (tid & 3) * 8) = av0;
    *(v8h*)(Alds + ((tid + 256) >> 2) * LDSP + (tid & 3) * 8) = av1;
#endif
#pragma unroll
    for (int j = 0; j < 8; ++j)                    // transpose B into [n][k]
      Blds[(bng * 8 + j) * LDSP + bkk] = bv[j];
    if (k0 + 32 < KDCN)
      __builtin_prefetch(Bg + (long)(k0 + 32 + bkk) * HW + p0 + bng * 8, 0, 1);
#if HAVE_TDM
    if (tid == 0) __builtin_amdgcn_s_wait_tensorcnt(0);
#endif
    __syncthreads();                               // A (TDM) + B visible to all waves
    v16h af[2], bf[2];
#pragma unroll
    for (int sm = 0; sm < 2; ++sm) {
      int ma = wm * 32 + sm * 16 + hl;
      v8h lo = *(const v8h*)(Alds + ma * LDSP + hi * 8);
      v8h up = *(const v8h*)(Alds + ma * LDSP + hi * 8 + 16);
      af[sm] = __builtin_shufflevector(lo, up, 0,1,2,3,4,5,6,7,8,9,10,11,12,13,14,15);
    }
#pragma unroll
    for (int sn = 0; sn < 2; ++sn) {
      int nb = wn * 32 + sn * 16 + hl;
      v8h lo = *(const v8h*)(Blds + nb * LDSP + hi * 8);
      v8h up = *(const v8h*)(Blds + nb * LDSP + hi * 8 + 16);
      bf[sn] = __builtin_shufflevector(lo, up, 0,1,2,3,4,5,6,7,8,9,10,11,12,13,14,15);
    }
#pragma unroll
    for (int sm = 0; sm < 2; ++sm)
#pragma unroll
      for (int sn = 0; sn < 2; ++sn)
        acc[sm][sn] = __builtin_amdgcn_wmma_f32_16x16x32_f16(
            false, af[sm], false, bf[sn], (short)0, acc[sm][sn], false, false);
  }
#pragma unroll
  for (int sm = 0; sm < 2; ++sm)
#pragma unroll
    for (int sn = 0; sn < 2; ++sn)
#pragma unroll
      for (int r = 0; r < 8; ++r) {
        int oc_r = m0 + wm * 32 + sm * 16 + r + 8 * hi;
        y[(long)(b * OCH + oc_r) * HW + p0 + wn * 32 + sn * 16 + hl] = acc[sm][sn][r];
      }
}

// ---------------- 5) ConvTranspose as 4 parity GEMMs: [256 x 1024] x [1024 x 4096] ----------------
__global__ void __launch_bounds__(256)
dl_gemm_up(const _Float16* __restrict__ A,       // [4][256][1024] f16
           const float* __restrict__ y,          // normalized y (d_out)
           float* __restrict__ yup) {            // [B][256][128][128] f32 (raw)
  __shared__ _Float16 Alds[128 * LDSP];
  __shared__ _Float16 Blds[64 * LDSP];
  int blk = blockIdx.x;
  int nt  = blk % 64;
  int mt  = (blk / 64) % 2;
  int par = (blk / 128) % 4;
  int b   = blk / 512;
  int p0 = nt * 64, m0 = mt * 128;
  int pa = par >> 1, pb = par & 1;
  int tid = threadIdx.x;
  int lane = tid & 31, wid = tid >> 5;
  int hl = lane & 15, hi = lane >> 4;
  int wm = wid & 3, wn = wid >> 2;
  v8f acc[2][2] = {};
  const _Float16* Ap = A + (long)par * OCH * KUP;
  int bkk = tid & 31, bng = tid >> 5;
#if HAVE_TDM
  unsigned ldsA = (unsigned)(unsigned long long)(const void*)&Alds[0];
#endif
  for (int k0 = 0; k0 < KUP; k0 += 32) {
#if !HAVE_TDM
    v8h av0 = *(const v8h*)(Ap + (long)(m0 + (tid >> 2)) * KUP + k0 + (tid & 3) * 8);
    v8h av1 = *(const v8h*)(Ap + (long)(m0 + ((tid + 256) >> 2)) * KUP + k0 + (tid & 3) * 8);
#endif
    int kr = k0 + bkk;
    int tau = kr >> 8, ic = kr & 255;
    int dh = pa - 1 + (tau >> 1), dw = pb - 1 + (tau & 1);
    const float* ysrc = y + (long)(b * CIN + ic) * HW;
    _Float16 tmp[8];
#pragma unroll
    for (int j = 0; j < 8; ++j) {                  // gather B elements (zero at borders)
      int q = p0 + bng * 8 + j;
      int ohh = q >> 6, oww = q & 63;
      int ih = ohh + dh, iw = oww + dw;
      float v = 0.f;
      if ((unsigned)ih < HH && (unsigned)iw < WW) v = ysrc[ih * WW + iw];
      tmp[j] = (_Float16)v;
    }
    __syncthreads();
#if HAVE_TDM
    if (tid == 0) tdm_load_a_tile(Ap + (long)m0 * KUP + k0, ldsA, KUP);
#else
    *(v8h*)(Alds + (tid >> 2) * LDSP + (tid & 3) * 8) = av0;
    *(v8h*)(Alds + ((tid + 256) >> 2) * LDSP + (tid & 3) * 8) = av1;
#endif
#pragma unroll
    for (int j = 0; j < 8; ++j) Blds[(bng * 8 + j) * LDSP + bkk] = tmp[j];
#if HAVE_TDM
    if (tid == 0) __builtin_amdgcn_s_wait_tensorcnt(0);
#endif
    __syncthreads();
    v16h af[2], bf[2];
#pragma unroll
    for (int sm = 0; sm < 2; ++sm) {
      int ma = wm * 32 + sm * 16 + hl;
      v8h lo = *(const v8h*)(Alds + ma * LDSP + hi * 8);
      v8h up = *(const v8h*)(Alds + ma * LDSP + hi * 8 + 16);
      af[sm] = __builtin_shufflevector(lo, up, 0,1,2,3,4,5,6,7,8,9,10,11,12,13,14,15);
    }
#pragma unroll
    for (int sn = 0; sn < 2; ++sn) {
      int nb = wn * 32 + sn * 16 + hl;
      v8h lo = *(const v8h*)(Blds + nb * LDSP + hi * 8);
      v8h up = *(const v8h*)(Blds + nb * LDSP + hi * 8 + 16);
      bf[sn] = __builtin_shufflevector(lo, up, 0,1,2,3,4,5,6,7,8,9,10,11,12,13,14,15);
    }
#pragma unroll
    for (int sm = 0; sm < 2; ++sm)
#pragma unroll
      for (int sn = 0; sn < 2; ++sn)
        acc[sm][sn] = __builtin_amdgcn_wmma_f32_16x16x32_f16(
            false, af[sm], false, bf[sn], (short)0, acc[sm][sn], false, false);
  }
#pragma unroll
  for (int sm = 0; sm < 2; ++sm)
#pragma unroll
    for (int sn = 0; sn < 2; ++sn)
#pragma unroll
      for (int r = 0; r < 8; ++r) {
        int oc = m0 + wm * 32 + sm * 16 + r + 8 * hi;
        int q = p0 + wn * 32 + sn * 16 + hl;
        int oh = 2 * (q >> 6) + pa, ow = 2 * (q & 63) + pb;
        yup[((long)(b * OCH + oc) * HO + oh) * WO + ow] = acc[sm][sn][r];
      }
}

// ---------------- 6) BatchNorm train-mode stats (biased var) + apply/ReLU ----------------
__global__ void dl_bn_stats(const float* __restrict__ src, int hw,
                            float* __restrict__ mean, float* __restrict__ inv) {
  int ch = blockIdx.x;
  __shared__ float ssum[256], ssq[256];
  float s = 0.f, q = 0.f;
  for (int b = 0; b < BATCH; ++b) {
    const float* p = src + (long)(b * OCH + ch) * hw;
    for (int i = threadIdx.x; i < hw; i += 256) { float v = p[i]; s += v; q += v * v; }
  }
  ssum[threadIdx.x] = s; ssq[threadIdx.x] = q;
  __syncthreads();
  for (int o = 128; o > 0; o >>= 1) {
    if (threadIdx.x < o) {
      ssum[threadIdx.x] += ssum[threadIdx.x + o];
      ssq[threadIdx.x]  += ssq[threadIdx.x + o];
    }
    __syncthreads();
  }
  if (threadIdx.x == 0) {
    float n = (float)(BATCH * hw);
    float m = ssum[0] / n;
    float v = ssq[0] / n - m * m;
    mean[ch] = m;
    inv[ch]  = rsqrtf(v + EPSBN);
  }
}

__global__ void dl_bn_apply(float* __restrict__ data, int hw,
                            const float* __restrict__ mean, const float* __restrict__ inv,
                            const float* __restrict__ gamma, const float* __restrict__ beta) {
  long g = (long)blockIdx.x * blockDim.x + threadIdx.x;
  long total = (long)BATCH * OCH * hw;
  if (g >= total) return;
  int ch = (int)((g / hw) % OCH);
  float v = data[g];
  v = (v - mean[ch]) * inv[ch] * gamma[ch] + beta[ch];
  data[g] = v > 0.f ? v : 0.f;
}

// ---------------- host-side orchestration ----------------
extern "C" void kernel_launch(void* const* d_in, const int* in_sizes, int n_in,
                              void* d_out, int out_size, void* d_ws, size_t ws_size,
                              hipStream_t stream) {
  (void)in_sizes; (void)n_in; (void)out_size; (void)ws_size;
  const float* x      = (const float*)d_in[0];
  const float* w_off  = (const float*)d_in[1];
  const float* b_off  = (const float*)d_in[2];
  const float* w_dcn  = (const float*)d_in[3];
  const float* gamma1 = (const float*)d_in[4];
  const float* beta1  = (const float*)d_in[5];
  const float* w_up   = (const float*)d_in[6];
  const float* gamma2 = (const float*)d_in[7];
  const float* beta2  = (const float*)d_in[8];

  float* y   = (float*)d_out;                         // [4,256,64,64]
  float* yup = y + (long)BATCH * OCH * HW;            // [4,256,128,128]

  char* base = (char*)d_ws;
  size_t off = 0;
  auto carve = [&](size_t bytes) -> void* {
    void* p = base + off;
    off = (off + bytes + 255) & ~(size_t)255;
    return p;
  };
  float*    om    = (float*)   carve((size_t)BATCH * 27 * HW * 4);
  int4*     ci    = (int4*)    carve((size_t)BATCH * 9 * HW * 16);
  float4*   cw    = (float4*)  carve((size_t)BATCH * 9 * HW * 16);
  _Float16* val   = (_Float16*)carve((size_t)BATCH * KDCN * HW * 2);   // 75.5 MB
  _Float16* Adcn  = (_Float16*)carve((size_t)OCH * KDCN * 2);
  _Float16* Aup   = (_Float16*)carve((size_t)4 * OCH * KUP * 2);
  float*    mean1 = (float*)   carve(OCH * 4);
  float*    inv1  = (float*)   carve(OCH * 4);
  float*    mean2 = (float*)   carve(OCH * 4);
  float*    inv2  = (float*)   carve(OCH * 4);

  dl_offset_conv<<<BATCH * 27 * HH, 64, 0, stream>>>(x, w_off, b_off, om);
  dl_coords<<<(BATCH * 9 * HW + 255) / 256, 256, 0, stream>>>(om, ci, cw);
  dl_sample<<<(int)(((long)BATCH * KDCN * HW) / 256), 256, 0, stream>>>(x, ci, cw, val);
  dl_cvt_wdcn<<<(OCH * KDCN + 255) / 256, 256, 0, stream>>>(w_dcn, Adcn);
  dl_gemm_dcn<<<BATCH * 2 * 64, 256, 0, stream>>>(Adcn, val, y);
  dl_bn_stats<<<OCH, 256, 0, stream>>>(y, HW, mean1, inv1);
  dl_bn_apply<<<(int)(((long)BATCH * OCH * HW + 255) / 256), 256, 0, stream>>>(
      y, HW, mean1, inv1, gamma1, beta1);
  dl_cvt_wup<<<(4 * OCH * KUP + 255) / 256, 256, 0, stream>>>(w_up, Aup);
  dl_gemm_up<<<BATCH * 4 * 2 * 64, 256, 0, stream>>>(Aup, y, yup);
  dl_bn_stats<<<OCH, 256, 0, stream>>>(yup, HO * WO, mean2, inv2);
  dl_bn_apply<<<(int)(((long)BATCH * OCH * HO * WO + 255) / 256), 256, 0, stream>>>(
      yup, HO * WO, mean2, inv2, gamma2, beta2);
}
